// LFS_59966333386838
// MI455X (gfx1250) — matrix operand order repaired and checked by reading, not compiled, not run on hardware
//
#include <hip/hip_runtime.h>
#include <hip/hip_bf16.h>

typedef _Float16 v16h __attribute__((ext_vector_type(16)));
typedef _Float16 v8h  __attribute__((ext_vector_type(8)));
typedef float    v8f  __attribute__((ext_vector_type(8)));
typedef unsigned int v4u __attribute__((ext_vector_type(4)));
typedef int          v4i __attribute__((ext_vector_type(4)));
typedef int          v8i __attribute__((ext_vector_type(8)));

#define WIN   10
#define HO    124
#define WO    124
#define NBAND 6
#define ROWS  4        // patch rows per workgroup (124 = 31 * 4)
#define KPAD  128      // K (patch pixels) padded 100 -> 128
#define NPAD  208      // N (freq Re/Im pairs) padded 200 -> 208
#define LDK   136      // padded basis row stride in halves (272B, 16B multiple)

#if __has_builtin(__builtin_amdgcn_tensor_load_to_lds) && __has_builtin(__builtin_amdgcn_s_wait_tensorcnt)
#define HAVE_TDM 1
#else
#define HAVE_TDM 0
#endif

// Band id of (unshifted) frequency bin g = 10*u + v, with fftshift + radial
// binning folded in. -1 for the 4 corner bins (r == 1.0 excluded by rr < edges[-1]).
__device__ __forceinline__ int band_of(int g) {
    int u = g / 10, v = g % 10;
    int su = (u + 5) % 10, sv = (v + 5) % 10;
    float cy = -1.0f + (2.0f / 9.0f) * (float)su;
    float cx = -1.0f + (2.0f / 9.0f) * (float)sv;
    float r = __builtin_amdgcn_sqrtf(cy * cy + cx * cx) * 0.7071067811865476f;
    if (r >= 1.0f) return -1;
    int b = (int)(r * 6.0f);
    return b > 5 ? 5 : b;
}

// lane-pair swap (xor 1) via DPP quad_perm(1,0,3,2) -- single VALU op
__device__ __forceinline__ float pair_swap(float v) {
    return __int_as_float(
        __builtin_amdgcn_mov_dpp(__float_as_int(v), 0xB1, 0xF, 0xF, true));
}

// TDM: async DMA of one 256(x) x 10(y) x 3(ch) f32 tile global -> LDS
__device__ __forceinline__ void tdm_load_rows(const float* gsrc, unsigned lds_off) {
#if HAVE_TDM
    unsigned long long ga = (unsigned long long)(uintptr_t)gsrc;
    v4u g0 = { 1u,                                    // count=1, user descriptor
               lds_off,                               // lds_addr (bytes)
               (unsigned)(ga & 0xFFFFFFFFu),          // global_addr[31:0]
               (unsigned)((ga >> 32) & 0x1FFFFFFu) | (2u << 30) }; // addr[56:32] | type=2
    v8i g1 = { (int)0x00020000,                       // data_size=2 (4 bytes), no multicast
               (int)(256u << 16),                     // tensor_dim0[15:0] << 16
               (int)(10u << 16),                      // dim0 hi16=0 | tensor_dim1 lo16
               (int)(256u << 16),                     // dim1 hi16=0 | tile_dim0=256
               (int)(10u | (3u << 16)),               // tile_dim1=10 | tile_dim2=3
               256,                                   // tensor_dim0_stride lo32
               0,                                     // stride0 hi16 | stride1 lo16 (65536: lo16=0)
               1 };                                   // tensor_dim1_stride[47:16] = 1
    v4i g2 = { 3, 1, 65536, 0 };                      // tensor_dim2=3, dim3=1, dim2_stride=65536
    v4i g3 = { 0, 0, 0, 0 };
#if defined(__clang_major__) && (__clang_major__ >= 23)
    v8i gz = { 0, 0, 0, 0, 0, 0, 0, 0 };
    __builtin_amdgcn_tensor_load_to_lds(g0, g1, g2, g3, gz, 0);
#else
    __builtin_amdgcn_tensor_load_to_lds(g0, g1, g2, g3, 0);
#endif
#else
    (void)gsrc; (void)lds_off;
#endif
}

__device__ __forceinline__ v16h load_b16(const _Float16* colp) {
    v8h lo = *(const v8h*)(colp);
    v8h hi = *(const v8h*)(colp + 16);
    return __builtin_shufflevector(lo, hi, 0,1,2,3,4,5,6,7,8,9,10,11,12,13,14,15);
}

__global__ __launch_bounds__(256)
void LFS_59966333386838_kernel(const float* __restrict__ xp, float* __restrict__ out) {
    // LDS (~145 KB; WGP budget is 320 KB)
    __shared__ _Float16 sBasis[NPAD][LDK];      // [n][k] FFT basis columns, f16
    __shared__ _Float16 sMask[16][LDK];         // [band][k] 0/1 masks, f16
    __shared__ float    sRaw[3][WIN][256];      // TDM landing buffer (raw channels)
    __shared__ float    sGray[2][WIN][256];     // double-buffered gray rows
    __shared__ _Float16 sAmpT[8][KPAD][16];     // per-wave amp, transposed [g][m]
    __shared__ float    sTrigC[10], sTrigS[10];
    __shared__ float    sInvCnt[8];

    const int tid  = threadIdx.x;
    const int lane = tid & 31;
    const int wave = tid >> 5;
    const int bimg = blockIdx.x / (HO / ROWS);
    const int h0   = (blockIdx.x % (HO / ROWS)) * ROWS;

    const float* imgsrc = xp + ((size_t)bimg * 3) * 65536;
    __builtin_prefetch(imgsrc + 2 * (size_t)h0 * 256, 0, 0);   // global_prefetch_b8

    // ---- phase A: kick off TDM for first row; trig table ----
#if HAVE_TDM
    const unsigned rawOff = (unsigned)(uintptr_t)(void*)&sRaw[0][0][0];
    if (tid < 32) tdm_load_rows(imgsrc + 2 * (size_t)h0 * 256, rawOff);
#endif
    if (tid < 10) {
        float ph = 0.6283185307179586f * (float)tid;  // 2*pi*t/10
        sTrigC[tid] = __builtin_cosf(ph);
        sTrigS[tid] = __builtin_sinf(ph);
    }
    __syncthreads();

    // ---- phase B: basis, mask, counts, zero amp tail (overlaps TDM) ----
    for (int i = tid; i < NPAD * KPAD; i += 256) {
        int n = i >> 7, k = i & 127;
        float val = 0.0f;
        if (n < 200 && k < 100) {
            int g = n >> 1;
            int u = g / 10, v = g % 10;
            int y = k / 10, x = k % 10;
            int q = (u * y + v * x) % 10;
            val = (n & 1) ? sTrigS[q] : sTrigC[q];
        }
        sBasis[n][k] = (_Float16)val;
    }
    for (int i = tid; i < 16 * KPAD; i += 256) {
        int n = i >> 7, k = i & 127;
        float val = 0.0f;
        if (n < NBAND && k < 100) val = (band_of(k) == n) ? 1.0f : 0.0f;
        sMask[n][k] = (_Float16)val;
    }
    if (tid < NBAND) {
        int cnt = 0;
        for (int g = 0; g < 100; ++g) cnt += (band_of(g) == tid) ? 1 : 0;
        sInvCnt[tid] = 1.0f / (float)(cnt < 1 ? 1 : cnt);
    }
    {   // zero only g in [104,128): never overwritten, read by GEMM2 every row
        unsigned* az = (unsigned*)&sAmpT[0][0][0];
        for (int i = tid; i < 8 * 24 * 8; i += 256) {
            int wv = i / 192, rem = i % 192;
            az[(wv * KPAD + 104) * 8 + rem] = 0u;
        }
    }

    // per-wave constants
    const int w0    = wave * 16;
    const int m     = lane & 15;
    const int half  = (lane < 16) ? 0 : 1;
    const int kbase = half ? 8 : 0;                 // A-layout K split
    const int boff  = half ? 8 : 0;                 // B-layout K split
    const int wclamp = (w0 + m < WO) ? (w0 + m) : (WO - 1);

    // ---- row loop: TDM(i+1) overlaps compute(i) ----
    for (int rr = 0; rr < ROWS; ++rr) {
        const int h   = h0 + rr;
        const int buf = rr & 1;
#if HAVE_TDM
        if (tid < 32) __builtin_amdgcn_s_wait_tensorcnt(0);
#else
        for (int i = tid; i < 3 * WIN * 256; i += 256) {
            int ch = i / (WIN * 256), rem = i % (WIN * 256);
            sRaw[ch][rem >> 8][rem & 255] =
                imgsrc[(size_t)ch * 65536 + (2 * h + (rem >> 8)) * 256 + (rem & 255)];
        }
#endif
        __syncthreads();                            // raw tile visible

        for (int i = tid; i < WIN * 256; i += 256) {  // raw -> gray (double buffer)
            int row = i >> 8, col = i & 255;
            sGray[buf][row][col] = 0.2989f * sRaw[0][row][col]
                                 + 0.587f  * sRaw[1][row][col]
                                 + 0.114f  * sRaw[2][row][col];
        }
        __syncthreads();                            // raw free, gray ready

#if HAVE_TDM
        if (rr + 1 < ROWS && tid < 32)              // prefetch next row's tile (async)
            tdm_load_rows(imgsrc + 2 * (size_t)(h + 1) * 256, rawOff);
#endif

        // A operand (patches) per documented 16-bit A layout
        v16h a[4];
#pragma unroll
        for (int s = 0; s < 4; ++s) {
            v16h av;
#pragma unroll
            for (int e = 0; e < 16; ++e) {
                int k = 32 * s + kbase + ((e < 8) ? e : (e + 8));
                float gval = 0.0f;
                if (k < 100) gval = sGray[buf][k / 10][2 * wclamp + (k % 10)];
                av[e] = (_Float16)gval;
            }
            a[s] = av;
        }

        // GEMM1 with double-buffered B operands (DS waits off critical path)
        v16h bb[4];
#pragma unroll
        for (int s = 0; s < 4; ++s) bb[s] = load_b16(&sBasis[m][32 * s + boff]);
        for (int j = 0; j < 13; ++j) {
            const int jn = (j < 12) ? (j + 1) : 12;
            v16h bn[4];
#pragma unroll
            for (int s = 0; s < 4; ++s)
                bn[s] = load_b16(&sBasis[16 * jn + m][32 * s + boff]);
            v8f c = {};
#pragma unroll
            for (int s = 0; s < 4; ++s)
                c = __builtin_amdgcn_wmma_f32_16x16x32_f16(
                        false, a[s], false, bb[s], (short)0, c, false, false);
            v8h amp8;
#pragma unroll
            for (int r = 0; r < 8; ++r) {
                float v = c[r];
                float u = pair_swap(v);             // Re<->Im partner lane (DPP)
                amp8[r] = (_Float16)__builtin_amdgcn_sqrtf(v * v + u * u);
            }
            if ((lane & 1) == 0) {                  // one b128 store per even lane
                int g = j * 8 + (m >> 1);
                *(v8h*)&sAmpT[wave][g][half * 8] = amp8;
            }
#pragma unroll
            for (int s = 0; s < 4; ++s) bb[s] = bn[s];
        }

        // GEMM2: band sums S[m][band] = amp[m][g] * Mask[g][band]
        v8f c2 = {};
#pragma unroll
        for (int s = 0; s < 4; ++s) {
            v16h a2;
#pragma unroll
            for (int e = 0; e < 16; ++e) {
                int k = 32 * s + kbase + ((e < 8) ? e : (e + 8));
                a2[e] = sAmpT[wave][k][m];
            }
            v16h b2 = load_b16(&sMask[m][32 * s + boff]);
            c2 = __builtin_amdgcn_wmma_f32_16x16x32_f16(
                     false, a2, false, b2, (short)0, c2, false, false);
        }

        // Epilogue: stat = 0.1 * S / count ; log10 = log2 * 0.30103 (v_log_f32)
        if (m < NBAND) {
            float scale = 0.1f * sInvCnt[m];
#pragma unroll
            for (int r = 0; r < 8; ++r) {
                int w = w0 + r + half * 8;
                if (w < WO) {
                    float stat = c2[r] * scale + 1e-6f;
                    out[(((size_t)bimg * NBAND + m) * HO + h) * WO + w] =
                        0.30102999566398f * __builtin_amdgcn_logf(stat);
                }
            }
        }
    }
}

extern "C" void kernel_launch(void* const* d_in, const int* in_sizes, int n_in,
                              void* d_out, int out_size, void* d_ws, size_t ws_size,
                              hipStream_t stream) {
    const float* x = (const float*)d_in[0];
    float* out = (float*)d_out;
    (void)in_sizes; (void)n_in; (void)d_ws; (void)ws_size; (void)out_size;
    dim3 grid(32 * (HO / ROWS));  // one workgroup per (image, 4-row group)
    dim3 block(256);              // 8 wave32 waves; wave i handles patches w=16i..16i+15
    LFS_59966333386838_kernel<<<grid, block, 0, stream>>>(x, out);
}
